// GCN_62732292325696
// MI455X (gfx1250) — matrix operand order repaired and checked by reading, not compiled, not run on hardware
//
#include <hip/hip_runtime.h>
#include <hip/hip_bf16.h>

// ---------------------------------------------------------------------------
// GNN message-passing for MI455X (gfx1250, wave32, WMMA).
// B=16, N=128, D=128, E=64, H=256, L=3.
// ---------------------------------------------------------------------------

constexpr int Bsz = 16;
constexpr int Nn  = 128;
constexpr int Dd  = 128;
constexpr int Ee  = 64;
constexpr int Hh  = 256;
constexpr int Ll  = 3;

typedef __attribute__((ext_vector_type(16))) __bf16 v16bf;
typedef __attribute__((ext_vector_type(8)))  float  v8f;

struct Frag32B { uint4 lo, hi; };

static __device__ __forceinline__ v16bf make_frag(uint4 lo, uint4 hi) {
    Frag32B f; f.lo = lo; f.hi = hi;
    return __builtin_bit_cast(v16bf, f);
}

// fp32 -> bf16 (round-to-nearest-even), stored as raw u16
static __device__ __forceinline__ unsigned short f2bf(float f) {
    unsigned u = __builtin_bit_cast(unsigned, f);
    u += 0x7FFFu + ((u >> 16) & 1u);
    return (unsigned short)(u >> 16);
}

static __device__ __forceinline__ float relu(float x) { return x > 0.f ? x : 0.f; }

// ---------------------------------------------------------------------------
// Prep kernels: bf16 conversions / transposes / initial masking
// ---------------------------------------------------------------------------

__global__ void gnn_prep_edges(const float* __restrict__ edges,
                               const unsigned char* __restrict__ mask,
                               unsigned short* __restrict__ ebf) {
    int idx = blockIdx.x * 256 + threadIdx.x;
    if (idx >= Bsz * Nn * Nn * Ee) return;
    int t = idx >> 6;          // drop e
    int j = t & (Nn - 1); t >>= 7;
    int i = t & (Nn - 1); int b = t >> 7;
    float m = (mask[b * Nn + i] && mask[b * Nn + j]) ? 1.f : 0.f;
    ebf[idx] = f2bf(edges[idx] * m);
}

__global__ void gnn_prep_x(const float* __restrict__ nodes,
                           const unsigned char* __restrict__ mask,
                           float* __restrict__ x0) {
    int idx = blockIdx.x * 256 + threadIdx.x;
    if (idx >= Bsz * Nn * Dd) return;
    int row = idx >> 7;  // b*N+i
    x0[idx] = nodes[idx] * (mask[row] ? 1.f : 0.f);
}

// w1eT[l][h][e] = bf16(msg_w1[l][D+e][h])
__global__ void gnn_prep_w1e(const float* __restrict__ msg_w1,
                             unsigned short* __restrict__ w1eT) {
    int idx = blockIdx.x * 256 + threadIdx.x;
    if (idx >= Ll * Hh * Ee) return;
    int e = idx & (Ee - 1); int t = idx >> 6;
    int h = t & (Hh - 1);   int l = t >> 8;
    w1eT[idx] = f2bf(msg_w1[((size_t)(l * (Dd + Ee) + Dd + e)) * Hh + h]);
}

// w2T[l][n][k] = bf16(msg_w2[l][k][n])
__global__ void gnn_prep_w2(const float* __restrict__ msg_w2,
                            unsigned short* __restrict__ w2T) {
    int idx = blockIdx.x * 256 + threadIdx.x;
    if (idx >= Ll * Hh * Hh) return;
    int k = idx & (Hh - 1); int t = idx >> 8;
    int n = t & (Hh - 1);   int l = t >> 8;
    w2T[idx] = f2bf(msg_w2[((size_t)(l * Hh + k)) * Hh + n]);
}

// ---------------------------------------------------------------------------
// node_part[b,i,h] = x[b,i,:] @ msg_w1[l,:D,h]   (f32 GEMV, tiny)
// ---------------------------------------------------------------------------
__global__ __launch_bounds__(256) void gnn_node_part(const float* __restrict__ x,
                                                     const float* __restrict__ w1n,
                                                     float* __restrict__ npart) {
    __shared__ float xs[Dd];
    int blk = blockIdx.x, tid = threadIdx.x;
    if (tid < Dd) xs[tid] = x[blk * Dd + tid];
    __syncthreads();
    float acc = 0.f;
    #pragma unroll 4
    for (int d = 0; d < Dd; ++d) acc += xs[d] * w1n[d * Hh + tid];
    npart[blk * Hh + tid] = acc;
}

// ---------------------------------------------------------------------------
// Fused edge-message kernel (WMMA bf16): one block per (b,i).
//   GEMM1: edge_part[128x256] = E[128x64] @ W1e[64x256]
//   h = relu(edge_part + node_part + b1)           (bf16 -> 64KB LDS)
//   GEMM2: msgs[128x256]      = h @ W2[256x256]
//   agg[h] = sum_j mask_j * msgs[j,h] + cnt*b2[h]  (masked column sum)
// ---------------------------------------------------------------------------
__global__ __launch_bounds__(256) void gnn_msg_wmma(
        const unsigned short* __restrict__ ebf,   // (B,N,N,E) bf16 (mask applied)
        const float* __restrict__ npart,          // (B,N,H)
        const float* __restrict__ bias1,          // (H)
        const float* __restrict__ bias2,          // (H)
        const unsigned short* __restrict__ w1eT,  // (H,E) bf16  (layer base)
        const unsigned short* __restrict__ w2T,   // (H,H) bf16  (layer base)
        const unsigned char* __restrict__ maskp,  // (B,N)
        float* __restrict__ agg)                  // (B,N,H)
{
    __shared__ __align__(16) unsigned short hbuf[Nn * Hh]; // 64KB; edge tile aliased
    const int blk  = blockIdx.x;            // b*N + i
    const int b    = blk >> 7;
    const int tid  = threadIdx.x;
    const int wave = tid >> 5;
    const int lane = tid & 31;
    const int nloc = lane & 15;
    const int half = lane >> 4;

    // ---- stage edge tile (128x64 bf16 = 16KB) into LDS --------------------
    unsigned short* etile = hbuf;
    {
        const uint4* src = (const uint4*)(ebf + (size_t)blk * (Nn * Ee));
        uint4* dst = (uint4*)etile;
        #pragma unroll
        for (int it = 0; it < (Nn * Ee / 8) / 256; ++it)
            dst[it * 256 + tid] = src[it * 256 + tid];
    }
    __syncthreads();

    v8f acc[8][2];
    const v8f vzero = {0.f, 0.f, 0.f, 0.f, 0.f, 0.f, 0.f, 0.f};
    #pragma unroll
    for (int rt = 0; rt < 8; ++rt) { acc[rt][0] = vzero; acc[rt][1] = vzero; }

    // ---- GEMM1: hoist 4 B-fragments of W1e^T, sweep row tiles -------------
    v16bf bw1[2][2]; // [col-tile local][k-step]
    #pragma unroll
    for (int ctl = 0; ctl < 2; ++ctl) {
        int h = (wave * 2 + ctl) * 16 + nloc;
        const unsigned short* wr = w1eT + h * Ee;
        #pragma unroll
        for (int ks = 0; ks < 2; ++ks) {
            int k = ks * 32 + half * 8;
            bw1[ctl][ks] = make_frag(*(const uint4*)(wr + k),
                                     *(const uint4*)(wr + k + 16));
        }
    }
    #pragma unroll
    for (int rt = 0; rt < 8; ++rt) {
        const unsigned short* erow = etile + (rt * 16 + nloc) * Ee;
        #pragma unroll
        for (int ks = 0; ks < 2; ++ks) {
            int k0 = ks * 32;
            v16bf a = make_frag(*(const uint4*)(erow + k0 + half * 8),
                                *(const uint4*)(erow + k0 + 16 + half * 8));
            acc[rt][0] = __builtin_amdgcn_wmma_f32_16x16x32_bf16(
                false, a, false, bw1[0][ks], (short)0, acc[rt][0], false, false);
            acc[rt][1] = __builtin_amdgcn_wmma_f32_16x16x32_bf16(
                false, a, false, bw1[1][ks], (short)0, acc[rt][1], false, false);
        }
    }
    __syncthreads();   // all LDS reads of the edge tile are complete

    // ---- epilogue 1: h = relu(edge_part + node_part + b1) -> bf16 LDS -----
    const int rowbase = blk * Hh;
    #pragma unroll
    for (int ctl = 0; ctl < 2; ++ctl) {
        int h = (wave * 2 + ctl) * 16 + nloc;
        float addv = npart[rowbase + h] + bias1[h];
        #pragma unroll
        for (int rt = 0; rt < 8; ++rt) {
            #pragma unroll
            for (int v = 0; v < 8; ++v) {
                int j = rt * 16 + half * 8 + v;   // C layout: lanes16-31 hold M=8..15
                hbuf[j * Hh + h] = f2bf(relu(acc[rt][ctl][v] + addv));
            }
        }
    }
    __syncthreads();

    // ---- GEMM2: msgs = h @ W2 (K=256, 8 k-steps) --------------------------
    #pragma unroll
    for (int rt = 0; rt < 8; ++rt) { acc[rt][0] = vzero; acc[rt][1] = vzero; }

    for (int ks = 0; ks < 8; ++ks) {
        int k0 = ks * 32;
        v16bf bw2a, bw2b;
        {
            int h0 = (wave * 2 + 0) * 16 + nloc;
            const unsigned short* wr = w2T + h0 * Hh + k0 + half * 8;
            bw2a = make_frag(*(const uint4*)wr, *(const uint4*)(wr + 16));
            int h1 = (wave * 2 + 1) * 16 + nloc;
            const unsigned short* wr1 = w2T + h1 * Hh + k0 + half * 8;
            bw2b = make_frag(*(const uint4*)wr1, *(const uint4*)(wr1 + 16));
        }
        #pragma unroll
        for (int rt = 0; rt < 8; ++rt) {
            const unsigned short* hrow = hbuf + (rt * 16 + nloc) * Hh;
            v16bf a = make_frag(*(const uint4*)(hrow + k0 + half * 8),
                                *(const uint4*)(hrow + k0 + 16 + half * 8));
            acc[rt][0] = __builtin_amdgcn_wmma_f32_16x16x32_bf16(
                false, a, false, bw2a, (short)0, acc[rt][0], false, false);
            acc[rt][1] = __builtin_amdgcn_wmma_f32_16x16x32_bf16(
                false, a, false, bw2b, (short)0, acc[rt][1], false, false);
        }
    }

    // ---- masked column sum over j, + cnt*b2, * mask_i ---------------------
    const unsigned char* mrow = maskp + b * Nn;
    float colsum0 = 0.f, colsum1 = 0.f;
    #pragma unroll
    for (int rt = 0; rt < 8; ++rt) {
        #pragma unroll
        for (int v = 0; v < 8; ++v) {
            int j = rt * 16 + half * 8 + v;
            float mj = mrow[j] ? 1.f : 0.f;
            colsum0 += acc[rt][0][v] * mj;
            colsum1 += acc[rt][1][v] * mj;
        }
    }
    colsum0 += __shfl_xor(colsum0, 16, 32);  // combine M=0..7 with M=8..15 halves
    colsum1 += __shfl_xor(colsum1, 16, 32);

    if (half == 0) {
        float cnt = 0.f;
        for (int j = 0; j < Nn; ++j) cnt += mrow[j] ? 1.f : 0.f;
        float mi = mrow[blk & (Nn - 1)] ? 1.f : 0.f;
        int h0 = (wave * 2 + 0) * 16 + nloc;
        int h1 = (wave * 2 + 1) * 16 + nloc;
        agg[rowbase + h0] = (colsum0 + cnt * bias2[h0]) * mi;
        agg[rowbase + h1] = (colsum1 + cnt * bias2[h1]) * mi;
    }
}

// ---------------------------------------------------------------------------
// Update MLP + residual + LayerNorm (per-row GEMV, small)
// ---------------------------------------------------------------------------
__global__ __launch_bounds__(256) void gnn_update(
        const float* __restrict__ x, const float* __restrict__ agg,
        const float* __restrict__ w1, const float* __restrict__ b1,
        const float* __restrict__ w2, const float* __restrict__ b2,
        const float* __restrict__ g,  const float* __restrict__ be,
        const unsigned char* __restrict__ maskp, float* __restrict__ xout)
{
    __shared__ float xs[Dd];
    __shared__ float as[Hh];
    __shared__ float u1[Hh];   // later reused for LN reduction
    __shared__ float ys[Dd];
    int blk = blockIdx.x, tid = threadIdx.x;
    if (tid < Dd) xs[tid] = x[blk * Dd + tid];
    as[tid] = agg[blk * Hh + tid];
    __syncthreads();

    float acc = b1[tid];
    #pragma unroll 4
    for (int d = 0; d < Dd; ++d) acc += xs[d] * w1[d * Hh + tid];
    #pragma unroll 4
    for (int k = 0; k < Hh; ++k) acc += as[k] * w1[(Dd + k) * Hh + tid];
    u1[tid] = relu(acc);
    __syncthreads();

    float mi = maskp[blk] ? 1.f : 0.f;
    if (tid < Dd) {
        float u = b2[tid];
        #pragma unroll 4
        for (int k = 0; k < Hh; ++k) u += u1[k] * w2[k * Dd + tid];
        ys[tid] = xs[tid] + u * mi;
    }
    __syncthreads();
    if (tid < Dd) { u1[tid] = ys[tid]; u1[Dd + tid] = ys[tid] * ys[tid]; }
    __syncthreads();
    for (int s = 64; s > 0; s >>= 1) {
        if (tid < s) { u1[tid] += u1[tid + s]; u1[Dd + tid] += u1[Dd + tid + s]; }
        __syncthreads();
    }
    float mu  = u1[0] * (1.f / Dd);
    float var = u1[Dd] * (1.f / Dd) - mu * mu;
    float inv = rsqrtf(var + 1e-5f);
    if (tid < Dd) {
        float yv = (ys[tid] - mu) * inv * g[tid] + be[tid];
        xout[blk * Dd + tid] = yv * mi;
    }
}

// ---------------------------------------------------------------------------
// Output MLP
// ---------------------------------------------------------------------------
__global__ __launch_bounds__(256) void gnn_out(
        const float* __restrict__ x,
        const float* __restrict__ w1, const float* __restrict__ b1,
        const float* __restrict__ w2, const float* __restrict__ b2,
        const unsigned char* __restrict__ maskp, float* __restrict__ out)
{
    __shared__ float xs[Dd];
    __shared__ float u1[Hh];
    int blk = blockIdx.x, tid = threadIdx.x;
    if (tid < Dd) xs[tid] = x[blk * Dd + tid];
    __syncthreads();
    float acc = b1[tid];
    #pragma unroll 4
    for (int d = 0; d < Dd; ++d) acc += xs[d] * w1[d * Hh + tid];
    u1[tid] = relu(acc);
    __syncthreads();
    float mi = maskp[blk] ? 1.f : 0.f;
    if (tid < Dd) {
        float u = b2[tid];
        #pragma unroll 4
        for (int k = 0; k < Hh; ++k) u += u1[k] * w2[k * Dd + tid];
        out[blk * Dd + tid] = u * mi;
    }
}

// ---------------------------------------------------------------------------
extern "C" void kernel_launch(void* const* d_in, const int* in_sizes, int n_in,
                              void* d_out, int out_size, void* d_ws, size_t ws_size,
                              hipStream_t stream) {
    (void)in_sizes; (void)n_in; (void)out_size; (void)ws_size;

    const float*         nodes  = (const float*)d_in[0];
    const float*         edges  = (const float*)d_in[1];
    const unsigned char* mask   = (const unsigned char*)d_in[2];
    const float*         msg_w1 = (const float*)d_in[3];
    const float*         msg_b1 = (const float*)d_in[4];
    const float*         msg_w2 = (const float*)d_in[5];
    const float*         msg_b2 = (const float*)d_in[6];
    const float*         upd_w1 = (const float*)d_in[7];
    const float*         upd_b1 = (const float*)d_in[8];
    const float*         upd_w2 = (const float*)d_in[9];
    const float*         upd_b2 = (const float*)d_in[10];
    const float*         ln_g   = (const float*)d_in[11];
    const float*         ln_b   = (const float*)d_in[12];
    const float*         out_w1 = (const float*)d_in[13];
    const float*         out_b1 = (const float*)d_in[14];
    const float*         out_w2 = (const float*)d_in[15];
    const float*         out_b2 = (const float*)d_in[16];

    // workspace carve-out (256B aligned)
    size_t off = 0;
    auto carve = [&](size_t bytes) {
        void* r = (char*)d_ws + off;
        off += (bytes + 255) & ~(size_t)255;
        return r;
    };
    unsigned short* ebf  = (unsigned short*)carve((size_t)Bsz * Nn * Nn * Ee * 2);
    float*          x0   = (float*)carve((size_t)Bsz * Nn * Dd * 4);
    float*          x1   = (float*)carve((size_t)Bsz * Nn * Dd * 4);
    float*          np   = (float*)carve((size_t)Bsz * Nn * Hh * 4);
    float*          agg  = (float*)carve((size_t)Bsz * Nn * Hh * 4);
    unsigned short* w1eT = (unsigned short*)carve((size_t)Ll * Hh * Ee * 2);
    unsigned short* w2T  = (unsigned short*)carve((size_t)Ll * Hh * Hh * 2);

    // prep
    gnn_prep_edges<<<(Bsz * Nn * Nn * Ee + 255) / 256, 256, 0, stream>>>(edges, mask, ebf);
    gnn_prep_x    <<<(Bsz * Nn * Dd + 255) / 256,      256, 0, stream>>>(nodes, mask, x0);
    gnn_prep_w1e  <<<(Ll * Hh * Ee + 255) / 256,       256, 0, stream>>>(msg_w1, w1eT);
    gnn_prep_w2   <<<(Ll * Hh * Hh + 255) / 256,       256, 0, stream>>>(msg_w2, w2T);

    float* xa = x0;
    float* xb = x1;
    const int rows = Bsz * Nn;  // 2048 blocks
    for (int l = 0; l < Ll; ++l) {
        gnn_node_part<<<rows, 256, 0, stream>>>(
            xa, msg_w1 + (size_t)l * (Dd + Ee) * Hh, np);
        gnn_msg_wmma<<<rows, 256, 0, stream>>>(
            ebf, np, msg_b1 + l * Hh, msg_b2 + l * Hh,
            w1eT + (size_t)l * Hh * Ee, w2T + (size_t)l * Hh * Hh, mask, agg);
        gnn_update<<<rows, 256, 0, stream>>>(
            xa, agg,
            upd_w1 + (size_t)l * (Dd + Hh) * Hh, upd_b1 + l * Hh,
            upd_w2 + (size_t)l * Hh * Dd,        upd_b2 + l * Dd,
            ln_g + l * Dd, ln_b + l * Dd, mask, xb);
        float* t = xa; xa = xb; xb = t;
    }
    gnn_out<<<rows, 256, 0, stream>>>(xa, out_w1, out_b1, out_w2, out_b2, mask,
                                      (float*)d_out);
}
